// SelfAttnFlex_15040975470631
// MI455X (gfx1250) — matrix operand order repaired and checked by reading, not compiled, not run on hardware
//
#include <hip/hip_runtime.h>
#include <hip/hip_bf16.h>

// SelfAttnFlex for MI455X (gfx1250, wave32, WMMA f16 16x16x32, f32 accum).
// cvt(f32->f16) -> {q,k GEMM + RoPE | v GEMM -> vT} -> per-wave flash attention -> out proj.
// Workspace layout (f16 elems): xh 2M | W1h 3M | W2h 1M | qh 2M | kh 2M | vT 2M | oh 2M = 28MB.

#define T_DIM 2048
#define E_DIM 1024
#define H_NUM 16
#define DH    64
#define PROW  80   // padded LDS row stride in f16 (160B) -> 16B-aligned b128 ds ops

typedef _Float16 h4  __attribute__((ext_vector_type(4)));
typedef _Float16 h8  __attribute__((ext_vector_type(8)));
typedef _Float16 h16 __attribute__((ext_vector_type(16)));
typedef float    f8  __attribute__((ext_vector_type(8)));

// ---------------- WMMA + operand-load helpers --------------------------

__device__ __forceinline__ f8 wmma16(h16 a, h16 b, f8 c) {
  // (neg_a, A, neg_b, B, c_mod, C, reuse_a, reuse_b)
  return __builtin_amdgcn_wmma_f32_16x16x32_f16(false, a, false, b, (short)0, c,
                                                false, false);
}

// A operand (16xK, lane l holds row l%16):
//   half 0 lanes: K = {0..7} and {16..23};  half 1 lanes: {8..15} and {24..31}
__device__ __forceinline__ h16 ld_a(const _Float16* rowp, int hs) {
  h8 lo = *(const h8*)(rowp + 8 * hs);
  h8 hi = *(const h8*)(rowp + 8 * hs + 16);
  return __builtin_shufflevector(lo, hi, 0,1,2,3,4,5,6,7,8,9,10,11,12,13,14,15);
}

// B operand (Kx16, lane l holds col l%16, column stored contiguously in K):
//   half 0 lanes: K = {0..7} and {16..23};  half 1 lanes: {8..15} and {24..31}
__device__ __forceinline__ h16 ld_b(const _Float16* colp, int hs) {
  h8 lo = *(const h8*)(colp + 8 * hs);
  h8 hi = *(const h8*)(colp + 8 * hs + 16);
  return __builtin_shufflevector(lo, hi, 0,1,2,3,4,5,6,7,8,9,10,11,12,13,14,15);
}

__device__ __forceinline__ h8 pack8(f8 v) {
  h8 r;
#pragma unroll
  for (int i = 0; i < 8; ++i) r[i] = (_Float16)v[i];
  return r;
}

// ---------------- f32 -> f16 conversion --------------------------------

__global__ __launch_bounds__(256) void cvt_k(const float* __restrict__ in,
                                             _Float16* __restrict__ out, int n4) {
  int i = blockIdx.x * blockDim.x + threadIdx.x;
  if (i < n4) {
    float4 v = ((const float4*)in)[i];
    h4 o;
    o[0] = (_Float16)v.x; o[1] = (_Float16)v.y;
    o[2] = (_Float16)v.z; o[3] = (_Float16)v.w;
    ((h4*)out)[i] = o;
  }
}

// ---------------- q/k GEMM + RoPE --------------------------------------
// grid.x = 32 (64-row slabs of W1: 0-15 q heads, 16-31 k heads)
// grid.y = 32, 4 waves/block, each wave one 16-wide t tile.
// M = W1 rows (one head's 64 dims), N = t  -> packed h8 stores into [h][t][d];
// RoPE pairs (d, d+32) live in accumulator tiles (0,2) and (1,3).
// Software-pipelined K loop: next k-step's 5 operand vectors are issued
// before the current step's WMMAs so load latency hides behind matrix ops.

__global__ __launch_bounds__(128) void qk_rope_k(
    const _Float16* __restrict__ xh, const _Float16* __restrict__ w1h,
    _Float16* __restrict__ qh, _Float16* __restrict__ kh) {
  const int lane = threadIdx.x & 31;
  const int wave = threadIdx.x >> 5;
  const int col  = lane & 15;
  const int hs   = lane >> 4;
  const int grp  = blockIdx.x;          // 64-row slab of W1 (q then k)
  const int type = grp >> 4;            // 0=q 1=k
  const int head = grp & 15;
  const int t0   = (blockIdx.y * 4 + wave) * 16;

  const _Float16* xrow = xh + (size_t)(t0 + col) * E_DIM;
  const _Float16* wb   = w1h + (size_t)(grp * 64 + col) * E_DIM;

  f8 a0 = {}, a1 = {}, a2 = {}, a3 = {};
  h16 bx  = ld_b(xrow, hs);
  h16 w0v = ld_a(wb + 0 * 16 * E_DIM, hs);
  h16 w1v = ld_a(wb + 1 * 16 * E_DIM, hs);
  h16 w2v = ld_a(wb + 2 * 16 * E_DIM, hs);
  h16 w3v = ld_a(wb + 3 * 16 * E_DIM, hs);
  for (int e0 = 0; e0 < E_DIM; e0 += 32) {
    const int en = (e0 + 32 < E_DIM) ? e0 + 32 : 0;  // branch-free tail
    h16 bxn  = ld_b(xrow + en, hs);
    h16 w0n  = ld_a(wb + 0 * 16 * E_DIM + en, hs);
    h16 w1n  = ld_a(wb + 1 * 16 * E_DIM + en, hs);
    h16 w2n  = ld_a(wb + 2 * 16 * E_DIM + en, hs);
    h16 w3n  = ld_a(wb + 3 * 16 * E_DIM + en, hs);
    a0 = wmma16(w0v, bx, a0);
    a1 = wmma16(w1v, bx, a1);
    a2 = wmma16(w2v, bx, a2);
    a3 = wmma16(w3v, bx, a3);
    bx = bxn; w0v = w0n; w1v = w1n; w2v = w2n; w3v = w3n;
  }

  // C layout: lane -> t = t0+col ; VGPR r of tile a -> d = 16a + r + 8*hs
  const float kf = -0.4152410118609203f;   // -log2(10000)/32
  const float tf = (float)(t0 + col);
#pragma unroll
  for (int r = 0; r < 8; ++r) {
    {  // pair tiles (0,2): d in [0,16)
      float d  = (float)(r + 8 * hs);
      float an = tf * exp2f(kf * d);
      float c = __cosf(an), s = __sinf(an);
      float lo = a0[r], hi = a2[r];
      a0[r] = lo * c - hi * s;
      a2[r] = hi * c + lo * s;
    }
    {  // pair tiles (1,3): d in [16,32)
      float d  = (float)(16 + r + 8 * hs);
      float an = tf * exp2f(kf * d);
      float c = __cosf(an), s = __sinf(an);
      float lo = a1[r], hi = a3[r];
      a1[r] = lo * c - hi * s;
      a3[r] = hi * c + lo * s;
    }
  }
  if (type == 0) {  // fold (1/sqrt(Dh)) * log2(e) into q -> exp2-domain softmax
    const float qs = 0.18033688011112042f;
#pragma unroll
    for (int r = 0; r < 8; ++r) {
      a0[r] *= qs; a1[r] *= qs; a2[r] *= qs; a3[r] *= qs;
    }
  }
  _Float16* dst = (type == 0 ? qh : kh) +
                  ((size_t)(head * T_DIM + t0 + col)) * DH + 8 * hs;
  *(h8*)(dst + 0)  = pack8(a0);
  *(h8*)(dst + 16) = pack8(a1);
  *(h8*)(dst + 32) = pack8(a2);
  *(h8*)(dst + 48) = pack8(a3);
}

// ---------------- v GEMM -> transposed vT[h][d][t] ----------------------
// grid.x = 16 heads, grid.y = 32, 4 waves/block.
// M = t, N = W1 rows -> D layout gives 8 consecutive t per lane -> packed
// h8 stores into vT so V is a contiguous B operand for P.V later.

__global__ __launch_bounds__(128) void v_k(
    const _Float16* __restrict__ xh, const _Float16* __restrict__ w1h,
    _Float16* __restrict__ vth) {
  const int lane = threadIdx.x & 31;
  const int wave = threadIdx.x >> 5;
  const int col  = lane & 15;
  const int hs   = lane >> 4;
  const int head = blockIdx.x;
  const int t0   = (blockIdx.y * 4 + wave) * 16;

  const _Float16* xrow = xh + (size_t)(t0 + col) * E_DIM;
  const _Float16* wb   = w1h + (size_t)(2048 + head * 64 + col) * E_DIM;

  f8 a0 = {}, a1 = {}, a2 = {}, a3 = {};
  h16 ax  = ld_a(xrow, hs);
  h16 w0v = ld_b(wb + 0 * 16 * E_DIM, hs);
  h16 w1v = ld_b(wb + 1 * 16 * E_DIM, hs);
  h16 w2v = ld_b(wb + 2 * 16 * E_DIM, hs);
  h16 w3v = ld_b(wb + 3 * 16 * E_DIM, hs);
  for (int e0 = 0; e0 < E_DIM; e0 += 32) {
    const int en = (e0 + 32 < E_DIM) ? e0 + 32 : 0;
    h16 axn = ld_a(xrow + en, hs);
    h16 w0n = ld_b(wb + 0 * 16 * E_DIM + en, hs);
    h16 w1n = ld_b(wb + 1 * 16 * E_DIM + en, hs);
    h16 w2n = ld_b(wb + 2 * 16 * E_DIM + en, hs);
    h16 w3n = ld_b(wb + 3 * 16 * E_DIM + en, hs);
    a0 = wmma16(ax, w0v, a0);
    a1 = wmma16(ax, w1v, a1);
    a2 = wmma16(ax, w2v, a2);
    a3 = wmma16(ax, w3v, a3);
    ax = axn; w0v = w0n; w1v = w1n; w2v = w2n; w3v = w3n;
  }
  // C layout: lane -> d col = 16a + col ; VGPR r -> t = t0 + r + 8*hs
  _Float16* dst = vth + ((size_t)(head * DH + col)) * T_DIM + t0 + 8 * hs;
  *(h8*)(dst + (size_t)0  * T_DIM) = pack8(a0);
  *(h8*)(dst + (size_t)16 * T_DIM) = pack8(a1);
  *(h8*)(dst + (size_t)32 * T_DIM) = pack8(a2);
  *(h8*)(dst + (size_t)48 * T_DIM) = pack8(a3);
}

// ---------------- flash attention (per-wave, causal) --------------------
// grid = (H, 32), 4 waves/block. Wave w of block y owns q-tile (y + 32w):
// causal work per tile grows with t0, so striding waves by 32 narrows the
// spread of per-workgroup s-block totals (52..80, vs 4..128 unstrided).

__global__ __launch_bounds__(128) void attn_k(
    const _Float16* __restrict__ qh, const _Float16* __restrict__ kh,
    const _Float16* __restrict__ vth, _Float16* __restrict__ oh) {
  __shared__ _Float16 smem[4 * 16 * PROW];
  const int lane = threadIdx.x & 31;
  const int wave = threadIdx.x >> 5;
  const int col  = lane & 15;
  const int hs   = lane >> 4;
  const int h    = blockIdx.x;
  const int t0   = (blockIdx.y + 32 * wave) * 16;   // load-balanced tile map
  _Float16* pl = smem + wave * 16 * PROW;

  // q tile resident in registers (A operand, 2 k-steps of 32)
  const _Float16* qb = qh + ((size_t)(h * T_DIM + t0 + col)) * DH;
  h16 qa0 = ld_a(qb + 0,  hs);
  h16 qa1 = ld_a(qb + 32, hs);

  f8 o0 = {}, o1 = {}, o2 = {}, o3 = {};
  float mrow[8], lrow[8];
#pragma unroll
  for (int r = 0; r < 8; ++r) { mrow[r] = -1e30f; lrow[r] = 0.f; }

  const _Float16* kb = kh + ((size_t)(h * T_DIM + col)) * DH;
  const _Float16* vb = vth + ((size_t)(h * DH + col)) * T_DIM;
  const int nblk = ((t0 + 15) >> 6) + 1;

  for (int b = 0; b < nblk; ++b) {
    const int s0 = b << 6;
    const _Float16* kp = kb + (size_t)s0 * DH;
    const _Float16* vp = vb + s0;
    // warm next s-block's K/V lines while this block computes
    __builtin_prefetch(kp + (size_t)64 * DH, 0, 3);
    __builtin_prefetch(vp + 64, 0, 3);

    // S = q . k^T : issue all 8 B-tile loads, then 8 WMMAs
    h16 k00 = ld_b(kp + (size_t)0  * DH + 0,  hs);
    h16 k10 = ld_b(kp + (size_t)16 * DH + 0,  hs);
    h16 k20 = ld_b(kp + (size_t)32 * DH + 0,  hs);
    h16 k30 = ld_b(kp + (size_t)48 * DH + 0,  hs);
    h16 k01 = ld_b(kp + (size_t)0  * DH + 32, hs);
    h16 k11 = ld_b(kp + (size_t)16 * DH + 32, hs);
    h16 k21 = ld_b(kp + (size_t)32 * DH + 32, hs);
    h16 k31 = ld_b(kp + (size_t)48 * DH + 32, hs);
    f8 s0a = {}, s1a = {}, s2a = {}, s3a = {};
    s0a = wmma16(qa0, k00, s0a);
    s1a = wmma16(qa0, k10, s1a);
    s2a = wmma16(qa0, k20, s2a);
    s3a = wmma16(qa0, k30, s3a);
    s0a = wmma16(qa1, k01, s0a);
    s1a = wmma16(qa1, k11, s1a);
    s2a = wmma16(qa1, k21, s2a);
    s3a = wmma16(qa1, k31, s3a);

    if (s0 + 63 > t0) {                // causal mask on diagonal blocks
#pragma unroll
      for (int r = 0; r < 8; ++r) {
        int t = t0 + r + 8 * hs;
        if (s0 + 0  + col > t) s0a[r] = -1e30f;
        if (s0 + 16 + col > t) s1a[r] = -1e30f;
        if (s0 + 32 + col > t) s2a[r] = -1e30f;
        if (s0 + 48 + col > t) s3a[r] = -1e30f;
      }
    }
    // online softmax (scores already in log2 domain; half-wave reductions
    // match the C-layout: lanes 0-15 / 16-31 hold disjoint row halves)
#pragma unroll
    for (int r = 0; r < 8; ++r) {
      float mx = fmaxf(fmaxf(s0a[r], s1a[r]), fmaxf(s2a[r], s3a[r]));
      mx = fmaxf(mx, __shfl_xor(mx, 1));
      mx = fmaxf(mx, __shfl_xor(mx, 2));
      mx = fmaxf(mx, __shfl_xor(mx, 4));
      mx = fmaxf(mx, __shfl_xor(mx, 8));
      float mnew  = fmaxf(mrow[r], mx);
      float alpha = exp2f(mrow[r] - mnew);
      mrow[r] = mnew;
      s0a[r] = exp2f(s0a[r] - mnew);
      s1a[r] = exp2f(s1a[r] - mnew);
      s2a[r] = exp2f(s2a[r] - mnew);
      s3a[r] = exp2f(s3a[r] - mnew);
      float rs = s0a[r] + s1a[r] + s2a[r] + s3a[r];
      rs += __shfl_xor(rs, 1); rs += __shfl_xor(rs, 2);
      rs += __shfl_xor(rs, 4); rs += __shfl_xor(rs, 8);
      lrow[r] = lrow[r] * alpha + rs;
      o0[r] *= alpha; o1[r] *= alpha; o2[r] *= alpha; o3[r] *= alpha;
      // P tile C-layout -> LDS row-major [t][s_local]
      int tr = (r + 8 * hs) * PROW;
      pl[tr + 0  + col] = (_Float16)s0a[r];
      pl[tr + 16 + col] = (_Float16)s1a[r];
      pl[tr + 32 + col] = (_Float16)s2a[r];
      pl[tr + 48 + col] = (_Float16)s3a[r];
    }
    // O += P . V   (A = P from LDS, B = vT rows, K=s contiguous):
    // issue both P loads and all 8 V loads, then 8 WMMAs
    h16 pa0 = ld_a(pl + col * PROW + 0,  hs);
    h16 pa1 = ld_a(pl + col * PROW + 32, hs);
    h16 v00 = ld_b(vp + (size_t)0  * T_DIM + 0,  hs);
    h16 v10 = ld_b(vp + (size_t)16 * T_DIM + 0,  hs);
    h16 v20 = ld_b(vp + (size_t)32 * T_DIM + 0,  hs);
    h16 v30 = ld_b(vp + (size_t)48 * T_DIM + 0,  hs);
    h16 v01 = ld_b(vp + (size_t)0  * T_DIM + 32, hs);
    h16 v11 = ld_b(vp + (size_t)16 * T_DIM + 32, hs);
    h16 v21 = ld_b(vp + (size_t)32 * T_DIM + 32, hs);
    h16 v31 = ld_b(vp + (size_t)48 * T_DIM + 32, hs);
    o0 = wmma16(pa0, v00, o0);
    o1 = wmma16(pa0, v10, o1);
    o2 = wmma16(pa0, v20, o2);
    o3 = wmma16(pa0, v30, o3);
    o0 = wmma16(pa1, v01, o0);
    o1 = wmma16(pa1, v11, o1);
    o2 = wmma16(pa1, v21, o2);
    o3 = wmma16(pa1, v31, o3);
  }
  // normalize, transpose through LDS, store o[t][e] row-major (f16)
#pragma unroll
  for (int r = 0; r < 8; ++r) {
    float inv = 1.0f / lrow[r];
    o0[r] *= inv; o1[r] *= inv; o2[r] *= inv; o3[r] *= inv;
    int tr = (r + 8 * hs) * PROW;
    pl[tr + 0  + col] = (_Float16)o0[r];
    pl[tr + 16 + col] = (_Float16)o1[r];
    pl[tr + 32 + col] = (_Float16)o2[r];
    pl[tr + 48 + col] = (_Float16)o3[r];
  }
#pragma unroll
  for (int it = 0; it < 4; ++it) {
    int idx = it * 32 + lane;      // 128 chunks of 8 f16
    int row = idx >> 3;
    int dc  = (idx & 7) * 8;
    h8 v = *(const h8*)(pl + row * PROW + dc);
    *(h8*)(oh + (size_t)(t0 + row) * E_DIM + h * DH + dc) = v;
  }
}

// ---------------- output projection: out = o @ W2^T (f32 out) -----------
// grid = (16 row-slabs of W2, 32), 4 waves/block. Same pipelined K loop.

__global__ __launch_bounds__(128) void proj_k(
    const _Float16* __restrict__ oh, const _Float16* __restrict__ w2h,
    float* __restrict__ out) {
  const int lane = threadIdx.x & 31;
  const int wave = threadIdx.x >> 5;
  const int col  = lane & 15;
  const int hs   = lane >> 4;
  const int i0   = blockIdx.x * 64;
  const int t0   = (blockIdx.y * 4 + wave) * 16;

  const _Float16* orow = oh + (size_t)(t0 + col) * E_DIM;
  const _Float16* wb   = w2h + (size_t)(i0 + col) * E_DIM;
  f8 a0 = {}, a1 = {}, a2 = {}, a3 = {};
  h16 bx  = ld_b(orow, hs);
  h16 w0v = ld_a(wb + 0 * 16 * E_DIM, hs);
  h16 w1v = ld_a(wb + 1 * 16 * E_DIM, hs);
  h16 w2v = ld_a(wb + 2 * 16 * E_DIM, hs);
  h16 w3v = ld_a(wb + 3 * 16 * E_DIM, hs);
  for (int e0 = 0; e0 < E_DIM; e0 += 32) {
    const int en = (e0 + 32 < E_DIM) ? e0 + 32 : 0;
    h16 bxn = ld_b(orow + en, hs);
    h16 w0n = ld_a(wb + 0 * 16 * E_DIM + en, hs);
    h16 w1n = ld_a(wb + 1 * 16 * E_DIM + en, hs);
    h16 w2n = ld_a(wb + 2 * 16 * E_DIM + en, hs);
    h16 w3n = ld_a(wb + 3 * 16 * E_DIM + en, hs);
    a0 = wmma16(w0v, bx, a0);
    a1 = wmma16(w1v, bx, a1);
    a2 = wmma16(w2v, bx, a2);
    a3 = wmma16(w3v, bx, a3);
    bx = bxn; w0v = w0n; w1v = w1n; w2v = w2n; w3v = w3n;
  }
  // C layout: lane -> t = t0+col ; VGPR r of tile a -> i = i0 + 16a + r + 8hs
  float* ob = out + (size_t)(t0 + col) * E_DIM + i0 + 8 * hs;
  *(f8*)(ob + 0)  = a0;
  *(f8*)(ob + 16) = a1;
  *(f8*)(ob + 32) = a2;
  *(f8*)(ob + 48) = a3;
}

// ---------------- launch ------------------------------------------------

extern "C" void kernel_launch(void* const* d_in, const int* in_sizes, int n_in,
                              void* d_out, int out_size, void* d_ws,
                              size_t ws_size, hipStream_t stream) {
  const float* x  = (const float*)d_in[0];
  const float* W1 = (const float*)d_in[1];
  const float* W2 = (const float*)d_in[2];
  float* out = (float*)d_out;

  _Float16* ws  = (_Float16*)d_ws;
  _Float16* xh  = ws;                                        // T*E
  _Float16* w1h = xh  + (size_t)T_DIM * E_DIM;               // 3E*E
  _Float16* w2h = w1h + (size_t)3 * E_DIM * E_DIM;           // E*E
  _Float16* qh  = w2h + (size_t)E_DIM * E_DIM;               // H*T*Dh
  _Float16* kh  = qh  + (size_t)T_DIM * E_DIM;               // H*T*Dh
  _Float16* vth = kh  + (size_t)T_DIM * E_DIM;               // H*Dh*T (transposed)
  _Float16* ohb = vth + (size_t)T_DIM * E_DIM;               // T*E

  int n4;
  n4 = T_DIM * E_DIM / 4;
  cvt_k<<<(n4 + 255) / 256, 256, 0, stream>>>(x, xh, n4);
  n4 = 3 * E_DIM * E_DIM / 4;
  cvt_k<<<(n4 + 255) / 256, 256, 0, stream>>>(W1, w1h, n4);
  n4 = E_DIM * E_DIM / 4;
  cvt_k<<<(n4 + 255) / 256, 256, 0, stream>>>(W2, w2h, n4);

  qk_rope_k<<<dim3(32, T_DIM / 64), 128, 0, stream>>>(xh, w1h, qh, kh);
  v_k<<<dim3(H_NUM, T_DIM / 64), 128, 0, stream>>>(xh, w1h, vth);
  attn_k<<<dim3(H_NUM, T_DIM / 64), 128, 0, stream>>>(qh, kh, vth, ohb);
  proj_k<<<dim3(E_DIM / 64, T_DIM / 64), 128, 0, stream>>>(ohb, w2h, out);
}
